// Gemma2Original_22883585753583
// MI455X (gfx1250) — matrix operand order repaired and checked by reading, not compiled, not run on hardware
//
#include <hip/hip_runtime.h>

typedef __attribute__((ext_vector_type(16))) _Float16 v16h;
typedef __attribute__((ext_vector_type(8)))  _Float16 v8h;
typedef __attribute__((ext_vector_type(4)))  _Float16 v4h;
typedef __attribute__((ext_vector_type(8)))  float    v8f;
typedef __attribute__((ext_vector_type(4)))  float    f4x;
typedef __attribute__((ext_vector_type(2)))  float    f2x;

#define ATT_B 2
#define ATT_H 16
#define ATT_S 2048
#define ATT_D 128
#define QTILE 128                     // 8 waves x 16 query rows
#define SCALE 0.08838834764831845f    // 1/sqrt(128)
#define CAP   50.0f

__device__ __forceinline__ v8f wmma_f16(v16h a, v16h b, v8f c) {
    return __builtin_amdgcn_wmma_f32_16x16x32_f16(
        false, a, false, b, (short)0, c, false, false);
}

// Fast soft-cap: CAP*tanh(s*SCALE/CAP) = CAP - 2*CAP/(exp(s*2*SCALE/CAP)+1)
__device__ __forceinline__ float softcap(float s) {
    const float t = __expf(s * (2.0f * SCALE / CAP));
    const float r = __builtin_amdgcn_rcpf(t + 1.0f);
    return fmaf(-2.0f * CAP, r, CAP);
}

__global__ __launch_bounds__(256)
void gemma2_attn_kernel(const float* __restrict__ q,
                        const float* __restrict__ k,   // [B,H,D,S]
                        const float* __restrict__ v,
                        float* __restrict__ out) {
    // ldsK[key][perm(d)] : perm places d-groups {0,16,8,24} so each A-fragment
    // is one contiguous 32B v16h.  ldsVt[d][key] : PV B-fragments contiguous.
    __shared__ __align__(32) _Float16 ldsK [32 * ATT_D];
    __shared__ __align__(32) _Float16 ldsVt[ATT_D * 32];

    const int S = ATT_S, D = ATT_D;
    const int qblocks = S / QTILE;
    const int bh = blockIdx.x / qblocks;
    const int qb = blockIdx.x % qblocks;
    const int q0 = qb * QTILE;

    const float* qbase = q + (size_t)bh * S * D;
    const float* kbase = k + (size_t)bh * D * S;
    const float* vbase = v + (size_t)bh * S * D;
    float*       obase = out + (size_t)bh * S * D;

    const int tid   = threadIdx.x;
    const int wave  = tid >> 5;
    const int lane  = tid & 31;
    const int lm    = lane & 15;
    const int lhalf = lane >> 4;

    // ---- Q as B-fragments (32d x 16m), loaded once (float4-vectorized) ----
    const int qg = q0 + wave * 16 + lm;         // this lane's query row
    const float* qrp = qbase + (size_t)qg * D;
    v16h bq[4];
    #pragma unroll
    for (int kc = 0; kc < 4; ++kc)
        #pragma unroll
        for (int i = 0; i < 4; ++i) {
            const f4x q4 = *(const f4x*)&qrp[kc * 32 + lhalf * 8 + (i >> 1) * 16 + (i & 1) * 4];
            #pragma unroll
            for (int e = 0; e < 4; ++e) bq[kc][4 * i + e] = (_Float16)q4[e];
        }

    // ---- Accumulators ----
    v8f oacc[8];
    #pragma unroll
    for (int i = 0; i < 8; ++i)
        #pragma unroll
        for (int j = 0; j < 8; ++j) oacc[i][j] = 0.0f;
    float mrun = -__builtin_inff();
    float lrun = 0.0f;

    const int kend  = q0 + QTILE;
    const int kcend = q0 + wave * 16 + 16;

    // ---- Software-pipelined staging: this thread's fixed patches ----
    // K patch: 2 keys (j0k, j0k+1) x 8 d (d0k..d0k+7); b64 loads along j.
    const int j0k = (tid & 15) * 2;
    const int d0k = (tid >> 4) * 8;
    const int gk  = (d0k >> 3) & 3;
    const int kst = (d0k & ~31) + ((gk & 1) * 16 + (gk >> 1) * 8);  // perm offset
    // V patch: 4 keys (j0v..+3) x 4 d (d0v..+3); b128 loads along d.
    const int d0v = (tid & 31) * 4;
    const int j0v = (tid >> 5) * 4;

    f2x kreg[8];
    f4x vreg[4];
    #pragma unroll
    for (int i = 0; i < 8; ++i)
        kreg[i] = *(const f2x*)&kbase[(size_t)(d0k + i) * S + j0k];
    #pragma unroll
    for (int i = 0; i < 4; ++i)
        vreg[i] = *(const f4x*)&vbase[(size_t)(j0v + i) * D + d0v];

    for (int kt = 0; kt < kend; kt += 32) {
        __syncthreads();               // LDS readers of previous tile done

        // ---- Convert staged registers -> LDS (f16) ----
        {
            v8h ra, rb;
            #pragma unroll
            for (int i = 0; i < 4; ++i) {
                const auto c0 = __builtin_amdgcn_cvt_pkrtz(kreg[2 * i][0], kreg[2 * i + 1][0]);
                ra[2 * i] = (_Float16)c0[0]; ra[2 * i + 1] = (_Float16)c0[1];
                const auto c1 = __builtin_amdgcn_cvt_pkrtz(kreg[2 * i][1], kreg[2 * i + 1][1]);
                rb[2 * i] = (_Float16)c1[0]; rb[2 * i + 1] = (_Float16)c1[1];
            }
            *(v8h*)&ldsK[j0k       * ATT_D + kst] = ra;
            *(v8h*)&ldsK[(j0k + 1) * ATT_D + kst] = rb;
        }
        #pragma unroll
        for (int i2 = 0; i2 < 4; ++i2) {
            const auto c0 = __builtin_amdgcn_cvt_pkrtz(vreg[0][i2], vreg[1][i2]);
            const auto c1 = __builtin_amdgcn_cvt_pkrtz(vreg[2][i2], vreg[3][i2]);
            v4h hv;
            hv[0] = (_Float16)c0[0]; hv[1] = (_Float16)c0[1];
            hv[2] = (_Float16)c1[0]; hv[3] = (_Float16)c1[1];
            *(v4h*)&ldsVt[(d0v + i2) * 32 + j0v] = hv;
        }

        // ---- Issue next tile's global loads (overlap with compute below) ----
        if (kt + 32 < kend) {
            #pragma unroll
            for (int i = 0; i < 8; ++i)
                kreg[i] = *(const f2x*)&kbase[(size_t)(d0k + i) * S + (kt + 32 + j0k)];
            #pragma unroll
            for (int i = 0; i < 4; ++i)
                vreg[i] = *(const f4x*)&vbase[(size_t)(kt + 32 + j0v + i) * D + d0v];
        }
        __syncthreads();               // tile visible to all waves

        if (kt >= kcend) continue;     // fully-masked for this wave's rows

        // ---- S^T tiles: A = K(16keys x 32d), B = Q(32d x 16m) ----
        v8f s0, s1;
        #pragma unroll
        for (int j = 0; j < 8; ++j) { s0[j] = 0.0f; s1[j] = 0.0f; }
        #pragma unroll
        for (int kc = 0; kc < 4; ++kc) {
            const v16h ak0 = *(const v16h*)&ldsK[lm        * ATT_D + kc * 32 + lhalf * 16];
            s0 = wmma_f16(ak0, bq[kc], s0);
            const v16h ak1 = *(const v16h*)&ldsK[(16 + lm) * ATT_D + kc * 32 + lhalf * 16];
            s1 = wmma_f16(ak1, bq[kc], s1);
        }

        // ---- Soft-cap + causal mask + online softmax (in-lane) ----
        float y0[8], y1[8];
        float mt = -__builtin_inff();
        #pragma unroll
        for (int r = 0; r < 8; ++r) {
            const int key0 = kt      + r + 8 * lhalf;
            const int key1 = kt + 16 + r + 8 * lhalf;
            float a = softcap(s0[r]);
            float b = softcap(s1[r]);
            if (key0 > qg) a = -__builtin_inff();
            if (key1 > qg) b = -__builtin_inff();
            y0[r] = a; y1[r] = b;
            mt = fmaxf(mt, fmaxf(a, b));
        }
        mt = fmaxf(mt, __shfl_xor(mt, 16, 32));
        const float mnew = fmaxf(mrun, mt);
        const float sc   = __expf(mrun - mnew);
        mrun = mnew;

        float p0[8], p1[8], rs = 0.0f;
        #pragma unroll
        for (int r = 0; r < 8; ++r) {
            p0[r] = __expf(y0[r] - mnew);
            p1[r] = __expf(y1[r] - mnew);
            rs += p0[r] + p1[r];
        }
        rs += __shfl_xor(rs, 16, 32);
        lrun = lrun * sc + rs;

        // ---- Rescale O only if the running max moved anywhere in this wave ----
        if (__ballot(sc != 1.0f)) {
            float scl[8];
            #pragma unroll
            for (int r = 0; r < 8; ++r) scl[r] = __shfl(sc, r + 8 * lhalf, 32);
            #pragma unroll
            for (int dt = 0; dt < 8; ++dt)
                #pragma unroll
                for (int r = 0; r < 8; ++r) oacc[dt][r] *= scl[r];
        }

        // ---- P -> A-fragment: in-lane register packing ----
        v16h ap;
        #pragma unroll
        for (int t = 0; t < 4; ++t) {
            const auto ca = __builtin_amdgcn_cvt_pkrtz(p0[2 * t], p0[2 * t + 1]);
            ap[2 * t]     = (_Float16)ca[0]; ap[2 * t + 1]     = (_Float16)ca[1];
            const auto cb = __builtin_amdgcn_cvt_pkrtz(p1[2 * t], p1[2 * t + 1]);
            ap[8 + 2 * t] = (_Float16)cb[0]; ap[8 + 2 * t + 1] = (_Float16)cb[1];
        }

        // ---- O += P(16m x 32keys) * V(32keys x 16d) per d-tile ----
        #pragma unroll
        for (int dt = 0; dt < 8; ++dt) {
            const v8h bv0 = *(const v8h*)&ldsVt[(dt * 16 + lm) * 32 + lhalf * 8];
            const v8h bv1 = *(const v8h*)&ldsVt[(dt * 16 + lm) * 32 + 16 + lhalf * 8];
            const v16h bv = __builtin_shufflevector(bv0, bv1, 0, 1, 2, 3, 4, 5, 6, 7, 8, 9, 10, 11, 12, 13, 14, 15);
            oacc[dt] = wmma_f16(ap, bv, oacc[dt]);
        }
    }

    // ---- Epilogue: normalize and store ----
    float inv[8];
    #pragma unroll
    for (int r = 0; r < 8; ++r)
        inv[r] = 1.0f / __shfl(lrun, r + 8 * lhalf, 32);
    #pragma unroll
    for (int dt = 0; dt < 8; ++dt)
        #pragma unroll
        for (int r = 0; r < 8; ++r) {
            const int rowg = q0 + wave * 16 + r + 8 * lhalf;
            obase[(size_t)rowg * D + dt * 16 + lm] = oacc[dt][r] * inv[r];
        }
}

extern "C" void kernel_launch(void* const* d_in, const int* in_sizes, int n_in,
                              void* d_out, int out_size, void* d_ws, size_t ws_size,
                              hipStream_t stream) {
    (void)in_sizes; (void)n_in; (void)out_size; (void)d_ws; (void)ws_size;
    const float* q = (const float*)d_in[0];
    const float* k = (const float*)d_in[1];
    const float* v = (const float*)d_in[2];
    float* out = (float*)d_out;

    const int blocks = ATT_B * ATT_H * (ATT_S / QTILE);   // 512
    gemma2_attn_kernel<<<dim3(blocks), dim3(256), 0, stream>>>(q, k, v, out);
}
